// MultiHeadCrossAttention_12369505813088
// MI455X (gfx1250) — matrix operand order repaired and checked by reading, not compiled, not run on hardware
//
#include <hip/hip_runtime.h>

// ---------------- types ----------------
typedef __bf16 bf16_t;
typedef __attribute__((ext_vector_type(16))) __bf16 v16bf;
typedef __attribute__((ext_vector_type(8)))  __bf16 v8bf;
typedef __attribute__((ext_vector_type(8)))  float  v8f;

#if __has_builtin(__builtin_amdgcn_tensor_load_to_lds)
#define USE_TDM 1
typedef __attribute__((ext_vector_type(4))) unsigned int u32x4;
typedef __attribute__((ext_vector_type(8))) int          i32x8;
typedef __attribute__((ext_vector_type(4))) int          i32x4;
typedef __attribute__((address_space(3))) char           lds_char;
#else
#define USE_TDM 0
#endif

static __device__ __forceinline__ bf16_t f2bf(float f) {
    unsigned int u = __builtin_bit_cast(unsigned int, f);
    unsigned int r = u + 0x7fffu + ((u >> 16) & 1u);   // round-to-nearest-even
    return __builtin_bit_cast(bf16_t, (unsigned short)(r >> 16));
}

// ---------------- fp32 -> bf16 convert ----------------
__global__ void cvt_f32_bf16(const float* __restrict__ in, bf16_t* __restrict__ out, int n) {
    int i = blockIdx.x * blockDim.x + threadIdx.x;
    int stride = gridDim.x * blockDim.x;
    for (; i < n; i += stride) out[i] = f2bf(in[i]);
}

// ---------------- WMMA GEMM ----------------
// C[M,N] = A[M,K](bf16 rowmajor) * B[K,N](bf16 rowmajor) + bias
// Block tile 128x64, BK=32. 256 threads = 8 waves, wave grid 4x2, 32x32 per wave.
// A tile staged via TDM (tensor_load_to_lds) when available; B tile staged
// transposed so B-fragments are contiguous 32B LDS loads.
// mode 0: store bf16 head-major [B,H,S,64]            (Q, K)
// mode 2: store bf16 head-major transposed [B,H,64,S] (V)
// mode 3: store fp32 [M,N]                            (output projection)
__global__ __launch_bounds__(256)
void gemm_bf16_wmma(const bf16_t* __restrict__ A, const bf16_t* __restrict__ Bw,
                    const float* __restrict__ bias, void* __restrict__ out,
                    int M, int N, int K, int mode, int sShift)
{
    __shared__ __align__(32) bf16_t As[128 * 32];   // row-major [row][k]
    __shared__ __align__(32) bf16_t Bs[64 * 32];    // transposed [col][k]

    const int tid  = threadIdx.x;
    const int wave = tid >> 5;
    const int lane = tid & 31;
    const int l    = lane & 15;
    const int hi   = lane >> 4;          // 0/1: which half-lane group
    const int wm   = wave >> 1;          // 0..3
    const int wn   = wave & 1;           // 0..1
    const int bm   = blockIdx.y * 128;
    const int bn   = blockIdx.x * 64;

    v8f acc[2][2];
#pragma unroll
    for (int i = 0; i < 2; ++i)
#pragma unroll
        for (int j = 0; j < 2; ++j)
#pragma unroll
            for (int e = 0; e < 8; ++e) acc[i][j][e] = 0.0f;

    for (int kk = 0; kk < K; kk += 32) {
        __syncthreads();
#if USE_TDM
        // ---- A tile via Tensor Data Mover: 2D tile 32(x) x 128(y), bf16 ----
        if (wave == 0) {
            unsigned long long ga =
                (unsigned long long)(const void*)A + ((unsigned long long)bm * K + kk) * 2ull;
            unsigned ldsAddr = (unsigned)(unsigned long long)(lds_char*)&As[0];
            u32x4 g0;
            g0[0] = 1u;                                   // count=1, user desc
            g0[1] = ldsAddr;                              // lds_addr[31:0]
            g0[2] = (unsigned)ga;                         // global_addr[31:0]
            g0[3] = (unsigned)((ga >> 32) & 0x01FFFFFFu)  // global_addr[56:32]
                    | (2u << 30);                         // type=2 ("image")
            i32x8 g1;
            g1[0] = 0x00010000;                           // data_size=1 (2 bytes)
            g1[1] = (int)(((unsigned)K & 0xFFFFu) << 16); // tensor_dim0[15:0]
            g1[2] = (int)(((unsigned)K >> 16) | (((unsigned)M & 0xFFFFu) << 16));
            g1[3] = (int)(((unsigned)M >> 16) | (32u << 16));   // tile_dim0=32
            g1[4] = 128;                                  // tile_dim1=128, tile_dim2=0
            g1[5] = K;                                    // tensor_dim0_stride[31:0]
            g1[6] = 0;
            g1[7] = 0;
            i32x4 gz4 = {0, 0, 0, 0};                     // groups 2/3 unused (2D)
            i32x8 gz8 = {0, 0, 0, 0, 0, 0, 0, 0};
            __builtin_amdgcn_tensor_load_to_lds(g0, g1, gz4, gz4, gz8, 0);
#if __has_builtin(__builtin_amdgcn_s_wait_tensorcnt)
            __builtin_amdgcn_s_wait_tensorcnt(0);
#else
            asm volatile("s_wait_tensorcnt 0x0" ::: "memory");
#endif
        }
#else
        {   // A tile: 128x32 = 4096 elems, 16 per thread (2x 16B loads)
            int idx = tid * 16;
            int r = idx >> 5, c = idx & 31;
            const bf16_t* src = A + (size_t)(bm + r) * K + kk + c;
            *(v8bf*)(&As[r * 32 + c])     = *(const v8bf*)(src);
            *(v8bf*)(&As[r * 32 + c + 8]) = *(const v8bf*)(src + 8);
        }
#endif
        {   // B tile: 32x64 = 2048 elems, 8 per thread, store transposed
            int idx = tid * 8;
            int r = idx >> 6, c = idx & 63;
            const bf16_t* src = Bw + (size_t)(kk + r) * N + bn + c;
            v8bf v = *(const v8bf*)src;
#pragma unroll
            for (int e = 0; e < 8; ++e) Bs[(c + e) * 32 + r] = v[e];
            if (kk + 32 < K)   // prefetch next K-step's B tile -> global_prefetch_b8
                __builtin_prefetch(src + (size_t)32 * N, 0, 1);
        }
        __syncthreads();

        // A fragments: lane l holds row, K = hi*16 + {0..15}
        v16bf afr[2], bfr[2];
#pragma unroll
        for (int t = 0; t < 2; ++t) {
            const bf16_t* p = &As[(wm * 32 + t * 16 + l) * 32];
            v8bf lo = *(const v8bf*)(p + hi * 16);
            v8bf hh = *(const v8bf*)(p + hi * 16 + 8);
#pragma unroll
            for (int e = 0; e < 8; ++e) { afr[t][e] = lo[e]; afr[t][8 + e] = hh[e]; }
        }
        // B fragments: lane l holds col, K = hi*16 + {0..15} (contiguous in Bs)
#pragma unroll
        for (int t = 0; t < 2; ++t) {
            bfr[t] = *(const v16bf*)(&Bs[(wn * 32 + t * 16 + l) * 32 + hi * 16]);
        }
#pragma unroll
        for (int i = 0; i < 2; ++i)
#pragma unroll
            for (int j = 0; j < 2; ++j)
                acc[i][j] = __builtin_amdgcn_wmma_f32_16x16x32_bf16(
                    false, afr[i], false, bfr[j], (short)0, acc[i][j], false, false);
    }

    // store: C layout = VGPR e -> row e+hi*8, lane l -> col
    const int sMask = (1 << sShift) - 1;
#pragma unroll
    for (int i = 0; i < 2; ++i)
#pragma unroll
        for (int j = 0; j < 2; ++j) {
            int n = bn + wn * 32 + j * 16 + l;
            float bsv = bias[n];
#pragma unroll
            for (int e = 0; e < 8; ++e) {
                int m = bm + wm * 32 + i * 16 + hi * 8 + e;
                float v = acc[i][j][e] + bsv;
                if (mode == 3) {
                    ((float*)out)[(size_t)m * N + n] = v;
                } else {
                    int b = m >> sShift, s = m & sMask;     // S is a power of two
                    int h = n >> 6, d = n & 63;
                    bf16_t bv = f2bf(v);
                    if (mode == 0)
                        ((bf16_t*)out)[((((size_t)(b * 16 + h)) << sShift) + s) * 64 + d] = bv;
                    else
                        ((bf16_t*)out)[((((size_t)(b * 16 + h)) * 64 + d) << sShift) + s] = bv;
                }
            }
        }
}

// ---------------- flash attention ----------------
// Qh,Kh: [B*H, S, 64] bf16.  VT: [B*H, 64, S] bf16.  ctx out: [B, SQ, 1024] bf16.
// 256 threads = 8 independent waves; each wave owns 16 query rows.
__global__ __launch_bounds__(256)
void attn_fa_wmma(const bf16_t* __restrict__ Qh, const bf16_t* __restrict__ Kh,
                  const bf16_t* __restrict__ VT, bf16_t* __restrict__ ctx,
                  int SQ, int SK)
{
    __shared__ __align__(32) bf16_t Pl[8][16 * 32];   // per-wave P scratch (16x32 bf16)

    const int tid  = threadIdx.x;
    const int wave = tid >> 5;
    const int lane = tid & 31;
    const int l    = lane & 15;
    const int hi   = lane >> 4;
    const int bh   = blockIdx.y;                       // b*16+h
    const int q0   = blockIdx.x * 128 + wave * 16;

    const size_t qbase = ((size_t)bh * SQ + q0) * 64;
    const size_t kbase = (size_t)bh * SK * 64;
    const size_t vbase = (size_t)bh * 64 * SK;

    // Q A-fragments, reused for the whole SK sweep (Dh=64 -> 2 frags of K=32)
    v16bf aq[2];
#pragma unroll
    for (int ka = 0; ka < 2; ++ka) {
        const bf16_t* p = Qh + qbase + (size_t)l * 64 + ka * 32;
        v8bf lo = *(const v8bf*)(p + hi * 16);
        v8bf hh = *(const v8bf*)(p + hi * 16 + 8);
#pragma unroll
        for (int e = 0; e < 8; ++e) { aq[ka][e] = lo[e]; aq[ka][8 + e] = hh[e]; }
    }

    v8f  cacc[4];
    float mrow[8], lrow[8];
#pragma unroll
    for (int t = 0; t < 4; ++t)
#pragma unroll
        for (int e = 0; e < 8; ++e) cacc[t][e] = 0.0f;
#pragma unroll
    for (int e = 0; e < 8; ++e) { mrow[e] = -1e30f; lrow[e] = 0.0f; }

    const float scale = 0.125f;          // 1/sqrt(64)
    const float LOG2E = 1.44269504f;

    for (int sk0 = 0; sk0 < SK; sk0 += 32) {
        // ---- scores: two 16x16 tiles over cols sk0..sk0+31 ----
        v8f s[2];
#pragma unroll
        for (int tj = 0; tj < 2; ++tj) {
            v8f z = {0.f, 0.f, 0.f, 0.f, 0.f, 0.f, 0.f, 0.f};
#pragma unroll
            for (int ka = 0; ka < 2; ++ka) {
                const bf16_t* kp = Kh + kbase + (size_t)(sk0 + tj * 16 + l) * 64
                                   + ka * 32 + hi * 16;
                v16bf bk = *(const v16bf*)kp;
                z = __builtin_amdgcn_wmma_f32_16x16x32_bf16(
                        false, aq[ka], false, bk, (short)0, z, false, false);
            }
            s[tj] = z;
        }
        // ---- online softmax (row reductions across 16-lane halves) ----
        float mnew[8], esc[8];
#pragma unroll
        for (int e = 0; e < 8; ++e) {
            float v0 = s[0][e] * scale, v1 = s[1][e] * scale;
            s[0][e] = v0; s[1][e] = v1;
            float mx = fmaxf(v0, v1);
            mx = fmaxf(mx, __shfl_xor(mx, 1));
            mx = fmaxf(mx, __shfl_xor(mx, 2));
            mx = fmaxf(mx, __shfl_xor(mx, 4));
            mx = fmaxf(mx, __shfl_xor(mx, 8));
            mnew[e] = fmaxf(mrow[e], mx);
            esc[e]  = exp2f((mrow[e] - mnew[e]) * LOG2E);
            mrow[e] = mnew[e];
        }
#pragma unroll
        for (int e = 0; e < 8; ++e) {
            float p0 = exp2f((s[0][e] - mrow[e]) * LOG2E);
            float p1 = exp2f((s[1][e] - mrow[e]) * LOG2E);
            float rs = p0 + p1;
            rs += __shfl_xor(rs, 1);
            rs += __shfl_xor(rs, 2);
            rs += __shfl_xor(rs, 4);
            rs += __shfl_xor(rs, 8);
            lrow[e] = lrow[e] * esc[e] + rs;
            int r = e + hi * 8;                        // C layout row
            Pl[wave][r * 32 + l]      = f2bf(p0);      // cols 0..15
            Pl[wave][r * 32 + 16 + l] = f2bf(p1);      // cols 16..31
        }
        // rescale running context
#pragma unroll
        for (int t = 0; t < 4; ++t)
#pragma unroll
            for (int e = 0; e < 8; ++e) cacc[t][e] *= esc[e];

        // P re-read in A-fragment layout (wave-local LDS, DS ops in-order per wave)
        v16bf ap;
        {
            const bf16_t* p = &Pl[wave][l * 32];
            v8bf lo = *(const v8bf*)(p + hi * 16);
            v8bf hh = *(const v8bf*)(p + hi * 16 + 8);
#pragma unroll
            for (int e = 0; e < 8; ++e) { ap[e] = lo[e]; ap[8 + e] = hh[e]; }
        }
        // ---- ctx += P @ V : 4 N-tiles of Dh, V^T rows contiguous ----
#pragma unroll
        for (int t = 0; t < 4; ++t) {
            const bf16_t* vp = VT + vbase + (size_t)(t * 16 + l) * SK + sk0 + hi * 16;
            v16bf bv = *(const v16bf*)vp;
            cacc[t] = __builtin_amdgcn_wmma_f32_16x16x32_bf16(
                          false, ap, false, bv, (short)0, cacc[t], false, false);
        }
    }

    // finalize + store merged-head ctx [B, SQ, 1024] bf16
    int b = bh >> 4, h = bh & 15;
#pragma unroll
    for (int t = 0; t < 4; ++t)
#pragma unroll
        for (int e = 0; e < 8; ++e) {
            float v = cacc[t][e] / lrow[e];
            int row = q0 + e + hi * 8;
            int col = h * 64 + t * 16 + l;
            ctx[((size_t)b * SQ + row) * 1024 + col] = f2bf(v);
        }
}

// ---------------- host launcher ----------------
extern "C" void kernel_launch(void* const* d_in, const int* in_sizes, int n_in,
                              void* d_out, int out_size, void* d_ws, size_t ws_size,
                              hipStream_t stream) {
    (void)in_sizes; (void)n_in; (void)out_size; (void)ws_size;
    const int B = 2, SQ = 2048, SK = 2048, D = 1024;
    const int M = B * SQ;                 // 4096 (== B*SK too)
    const int S_SHIFT = 11;               // log2(2048)

    const float* x1 = (const float*)d_in[0];
    const float* x2 = (const float*)d_in[1];
    const float* Wq = (const float*)d_in[2]; const float* bq = (const float*)d_in[3];
    const float* Wk = (const float*)d_in[4]; const float* bk = (const float*)d_in[5];
    const float* Wv = (const float*)d_in[6]; const float* bv = (const float*)d_in[7];
    const float* Wo = (const float*)d_in[8]; const float* bo = (const float*)d_in[9];

    // workspace carve-up (bf16 buffers)
    char* ws = (char*)d_ws;
    const size_t szX = (size_t)M * D * sizeof(bf16_t);     // 8 MB
    const size_t szW = (size_t)D * D * sizeof(bf16_t);     // 2 MB
    bf16_t* xb1 = (bf16_t*)(ws);            ws += szX;
    bf16_t* xb2 = (bf16_t*)(ws);            ws += szX;
    bf16_t* wqb = (bf16_t*)(ws);            ws += szW;
    bf16_t* wkb = (bf16_t*)(ws);            ws += szW;
    bf16_t* wvb = (bf16_t*)(ws);            ws += szW;
    bf16_t* wob = (bf16_t*)(ws);            ws += szW;
    bf16_t* Qh  = (bf16_t*)(ws);            ws += szX;     // [B*H, SQ, 64]
    bf16_t* Kh  = (bf16_t*)(ws);            ws += szX;     // [B*H, SK, 64]
    bf16_t* VT  = (bf16_t*)(ws);            ws += szX;     // [B*H, 64, SK]
    bf16_t* ctx = (bf16_t*)(ws);            ws += szX;     // [B, SQ, 1024]

    // 1) convert inputs to bf16
    cvt_f32_bf16<<<4096, 256, 0, stream>>>(x1, xb1, M * D);
    cvt_f32_bf16<<<4096, 256, 0, stream>>>(x2, xb2, M * D);
    cvt_f32_bf16<<<2048, 256, 0, stream>>>(Wq, wqb, D * D);
    cvt_f32_bf16<<<2048, 256, 0, stream>>>(Wk, wkb, D * D);
    cvt_f32_bf16<<<2048, 256, 0, stream>>>(Wv, wvb, D * D);
    cvt_f32_bf16<<<2048, 256, 0, stream>>>(Wo, wob, D * D);

    dim3 ggrid(D / 64, M / 128);           // (16, 32)
    // 2) projections
    gemm_bf16_wmma<<<ggrid, 256, 0, stream>>>(xb1, wqb, bq, Qh, M, D, D, 0, S_SHIFT);
    gemm_bf16_wmma<<<ggrid, 256, 0, stream>>>(xb2, wkb, bk, Kh, M, D, D, 0, S_SHIFT);
    gemm_bf16_wmma<<<ggrid, 256, 0, stream>>>(xb2, wvb, bv, VT, M, D, D, 2, S_SHIFT);
    // 3) attention
    dim3 agrid(SQ / 128, B * 16);          // (16, 32)
    attn_fa_wmma<<<agrid, 256, 0, stream>>>(Qh, Kh, VT, ctx, SQ, SK);
    // 4) output projection -> fp32
    gemm_bf16_wmma<<<ggrid, 256, 0, stream>>>(ctx, wob, bo, d_out, M, D, D, 3, S_SHIFT);
}
